// SAM_77764677861692
// MI455X (gfx1250) — compile-verified, matrix operand order checked by reading, not run on hardware
//
#include <hip/hip_runtime.h>
#include <hip/hip_bf16.h>

// ---------------------------------------------------------------------------
// MI455X (gfx1250) fused stereo cross-attention pipeline, round 3.
// - All GEMM work on v_wmma_f32_16x16x32_f16 (wave32, f32 accumulate).
// - 3x3 convs: implicit GEMM over zero-padded NHWC f16 input. Fully unrolled
//   72-step K loop; A tile (32px x 32K) double-buffered in LDS via
//   global_load_async_to_lds_b64 (ASYNCcnt) issued by all 256 threads;
//   4 accumulators per wave (2 oc tiles x 2 pixel halves), next-step weight
//   fragments preloaded into registers to hide latency across the barrier.
// - Attention score/softmax/apply fused per (b,h) scanline in LDS.
// Workspace requirement: ~880 MB.
// ---------------------------------------------------------------------------

#define Bsz   8
#define Cch   256
#define Hh    160
#define Ww    160
#define HP    162               // padded spatial (1-px zero halo)
#define WP    162
#define NPX   (Bsz*Hh*Ww)       // 204800 pixels
#define K3    (9*Cch)           // 2304 im2col depth
#define KBN   544               // 513 padded to multiple of 32
#define WPAD  164               // LDS row pitch for 160-wide score

#ifndef USE_ASYNC_LDS
#define USE_ASYNC_LDS 1
#endif

typedef __attribute__((ext_vector_type(16))) _Float16 v16h;
typedef __attribute__((ext_vector_type(8)))  _Float16 v8h;
typedef __attribute__((ext_vector_type(4)))  _Float16 v4h;
typedef __attribute__((ext_vector_type(8)))  float    v8f;

__device__ __forceinline__ v8f wmma16(v16h a, v16h b, v8f c) {
    return __builtin_amdgcn_wmma_f32_16x16x32_f16(
        false, a, false, b, (short)0, c, false, false);
}

__device__ __forceinline__ v16h frag_ld(const _Float16* p) {
    // lane fragment: elements 0..7 at p[0..7], 8..15 at p[16..23]
    v8h lo = *(const v8h*)p;
    v8h hi = *(const v8h*)(p + 16);
    v16h r;
    for (int i = 0; i < 8; ++i) { r[i] = lo[i]; r[8 + i] = hi[i]; }
    return r;
}

// 8B global -> LDS, asynchronous (ASYNCcnt) when enabled
__device__ __forceinline__ void stage_b64(const _Float16* g, _Float16* l) {
#if USE_ASYNC_LDS
    unsigned off = (unsigned)(unsigned long long)l;   // LDS byte offset
    asm volatile("global_load_async_to_lds_b64 %0, %1, off"
                 :: "v"(off), "v"(g) : "memory");
#else
    *(v4h*)l = *(const v4h*)g;
#endif
}
__device__ __forceinline__ void stage_wait() {
#if USE_ASYNC_LDS
    asm volatile("s_wait_asynccnt 0x0" ::: "memory");
#endif
}

// ------------------------------ pack kernels -------------------------------

__global__ __launch_bounds__(256) void k_pack_x(const float* __restrict__ x,
                                                _Float16* __restrict__ xh,
                                                _Float16* __restrict__ xpad) {
    long i = (long)blockIdx.x * 256 + threadIdx.x;        // p*C + c
    int  c = (int)(i & (Cch - 1));
    long p = i >> 8;
    int  w = (int)(p % Ww);
    int  h = (int)((p / Ww) % Hh);
    int  b = (int)(p / ((long)Ww * Hh));
    _Float16 v = (_Float16)x[(((long)b * Cch + c) * Hh + h) * Ww + w];
    xh[i] = v;
    xpad[(((long)b * HP + h + 1) * WP + (w + 1)) * Cch + c] = v;
}

__global__ __launch_bounds__(256) void k_pack_w3(const float* __restrict__ w,
                                                 _Float16* __restrict__ wT) {
    long i = (long)blockIdx.x * 256 + threadIdx.x;        // oc*K3 + k
    int  k  = (int)(i % K3);
    int  oc = (int)(i / K3);
    int  tap = k >> 8;                                    // dy*3+dx
    int  c   = k & (Cch - 1);
    wT[i] = (_Float16)w[(long)oc * Cch * 9 + (long)c * 9 + tap];
}

__global__ __launch_bounds__(256) void k_pack_w1(const float* __restrict__ w,
                                                 _Float16* __restrict__ wT) {
    long i = (long)blockIdx.x * 256 + threadIdx.x;        // (out,in) both
    wT[i] = (_Float16)w[i];
}

__global__ __launch_bounds__(256) void k_pack_bn(const float* __restrict__ w,
                                                 _Float16* __restrict__ wT) {
    long i = (long)blockIdx.x * 256 + threadIdx.x;        // oc*KBN + k
    int  k  = (int)(i % KBN);
    int  oc = (int)(i / KBN);
    wT[i] = (k < 513) ? (_Float16)w[(long)oc * 513 + k] : (_Float16)0.f;
}

// --------------------- implicit-GEMM 3x3 conv (WMMA) -----------------------
// Block: 32 pixels x 256 oc, 8 waves. Wave wv owns oc tiles [wv*32, wv*32+32);
// 4 accumulators = 2 oc tiles x 2 pixel halves. A tile (32x32) staged in LDS
// (double buffered, async b64 by all 256 threads), B streamed from global
// with next-step register preload. Fully unrolled 72 K-steps.

__global__ __launch_bounds__(256) void k_conv3(const _Float16* __restrict__ xpad,
                                               const _Float16* __restrict__ wT,
                                               const _Float16* __restrict__ resid,
                                               _Float16* __restrict__ out,
                                               int mode, int outPadded) {
    __shared__ _Float16 Abuf[2][32][40];                  // 80B pitch, 5KB
    int tid = threadIdx.x;
    int lane = tid & 31, wv = tid >> 5;
    int hl = lane >> 4, ln = lane & 15;

    long pb  = (long)blockIdx.x * 32;
    int  ocb = wv * 32;

    // staging coords: 32 rows x 8 x 8B segments = 256 threads, no divergence
    int sr = tid >> 3, sseg = tid & 7;
    long sp = pb + sr;
    int sw = (int)(sp % Ww);
    int sh = (int)((sp / Ww) % Hh);
    int sb = (int)(sp / ((long)Ww * Hh));
    const _Float16* sbase =
        xpad + (((long)sb * HP + sh + 1) * WP + (sw + 1)) * Cch + sseg * 4;
    _Float16* sdst[2] = { &Abuf[0][sr][sseg * 4], &Abuf[1][sr][sseg * 4] };

    const _Float16* wrow0 = wT + (long)(ocb + ln) * K3 + hl * 8;
    const _Float16* wrow1 = wT + (long)(ocb + 16 + ln) * K3 + hl * 8;

    v8f acc00 = {0.f,0.f,0.f,0.f,0.f,0.f,0.f,0.f};
    v8f acc01 = acc00, acc10 = acc00, acc11 = acc00;

    auto stage = [&](int ks) {                    // ks constant after unroll
        int tap = ks >> 3;
        int c0  = (ks & 7) * 32;
        int toff = ((tap / 3 - 1) * WP + (tap % 3 - 1)) * Cch;
        stage_b64(sbase + toff + c0, sdst[ks & 1]);
    };

    stage(0);
    stage_wait();
    __syncthreads();

    v16h bf0 = frag_ld(wrow0);
    v16h bf1 = frag_ld(wrow1);

#pragma unroll
    for (int ks = 0; ks < 72; ++ks) {
        v16h nb0, nb1;
        if (ks + 1 < 72) {
            stage(ks + 1);
            __builtin_prefetch(wrow0 + (ks + 2) * 32, 0, 3);
            nb0 = frag_ld(wrow0 + (ks + 1) * 32);
            nb1 = frag_ld(wrow1 + (ks + 1) * 32);
        }
        int buf = ks & 1;
        v16h a0 = frag_ld(&Abuf[buf][ln][hl * 8]);
        v16h a1 = frag_ld(&Abuf[buf][16 + ln][hl * 8]);
        acc00 = wmma16(a0, bf0, acc00);
        acc10 = wmma16(a1, bf0, acc10);
        acc01 = wmma16(a0, bf1, acc01);
        acc11 = wmma16(a1, bf1, acc11);

        stage_wait();                             // next-buffer loads done
        __syncthreads();                          // all reads of buf done
        bf0 = nb0; bf1 = nb1;
    }

    for (int ot = 0; ot < 2; ++ot) {
        int oc = ocb + ot * 16 + ln;
        for (int mh = 0; mh < 2; ++mh) {
            v8f a = ot ? (mh ? acc11 : acc01) : (mh ? acc10 : acc00);
            for (int r = 0; r < 8; ++r) {
                long p = pb + mh * 16 + r + hl * 8;
                float v = a[r];
                if (mode == 0) v = v / (1.f + __expf(-v));        // SiLU
                else           v += (float)resid[p * Cch + oc];   // + skip
                if (outPadded) {
                    int w = (int)(p % Ww);
                    int h = (int)((p / Ww) % Hh);
                    int b = (int)(p / ((long)Ww * Hh));
                    out[(((long)b * HP + h + 1) * WP + (w + 1)) * Cch + oc] =
                        (_Float16)v;
                } else {
                    out[p * Cch + oc] = (_Float16)v;
                }
            }
        }
    }
}

// ------------------------ generic K=256 1x1 GEMM ---------------------------

__global__ __launch_bounds__(256) void k_gemm256(const _Float16* __restrict__ xin,
                                                 const _Float16* __restrict__ wT,
                                                 const float* __restrict__ bias,
                                                 _Float16* __restrict__ out) {
    int lane = threadIdx.x & 31, wv = threadIdx.x >> 5;
    int tile = blockIdx.x * 8 + wv;
    int mt = tile >> 4, nt = tile & 15;
    long pbase = (long)mt * 16;
    int  ocbase = nt * 16;
    int  hl = lane >> 4, ln = lane & 15;

    const _Float16* arow = xin + (pbase + ln) * Cch;
    const _Float16* brow = wT + (long)(ocbase + ln) * Cch;
    v8f acc = {0.f,0.f,0.f,0.f,0.f,0.f,0.f,0.f};

#pragma unroll
    for (int kk = 0; kk < 8; ++kk) {
        int k0 = kk * 32 + hl * 8;
        acc = wmma16(frag_ld(arow + k0), frag_ld(brow + k0), acc);
    }
    float bb = bias[ocbase + ln];
    for (int r = 0; r < 8; ++r) {
        long pp = pbase + r + hl * 8;
        out[pp * Cch + ocbase + ln] = (_Float16)(acc[r] + bb);
    }
}

// ---------------- attention: score + softmax + apply, per (b,h) ------------

__global__ __launch_bounds__(256) void k_attn(const _Float16* __restrict__ Qh,
                                              const _Float16* __restrict__ Sh,
                                              const _Float16* __restrict__ xlh,
                                              const _Float16* __restrict__ xrh,
                                              _Float16* __restrict__ bufL,
                                              _Float16* __restrict__ bufR,
                                              float* __restrict__ mvalL,
                                              float* __restrict__ mvalR) {
    extern __shared__ char smem[];
    float* sc   = (float*)smem;                 // 160 x WPAD score
    float* rmax = sc + 160 * WPAD;
    float* rsum = rmax + 160;
    float* cmax = rsum + 160;
    float* csum = cmax + 160;

    int bh = blockIdx.x;                        // b*H + h
    long rowbase = (long)bh * Ww;
    const _Float16* Qb = Qh + rowbase * Cch;
    const _Float16* Sb = Sh + rowbase * Cch;
    int lane = threadIdx.x & 31, wv = threadIdx.x >> 5;
    int hl = lane >> 4, ln = lane & 15;

    // ---- score = Q(160x256) x S^T into LDS ----
    for (int t = wv; t < 100; t += 8) {
        int mt = t / 10, nt = t % 10;
        const _Float16* arow = Qb + (long)(mt * 16 + ln) * Cch;
        const _Float16* brow = Sb + (long)(nt * 16 + ln) * Cch;
        v8f acc = {0.f,0.f,0.f,0.f,0.f,0.f,0.f,0.f};
#pragma unroll
        for (int kk = 0; kk < 8; ++kk) {
            int k0 = kk * 32 + hl * 8;
            acc = wmma16(frag_ld(arow + k0), frag_ld(brow + k0), acc);
        }
        for (int r = 0; r < 8; ++r) {
            int m = r + hl * 8;
            sc[(mt * 16 + m) * WPAD + nt * 16 + ln] = acc[r];
        }
    }
    __syncthreads();

    // ---- row / column softmax statistics ----
    for (int i = threadIdx.x; i < Ww; i += 256) {
        float mx = -3.4e38f;
        for (int j = 0; j < Ww; ++j) mx = fmaxf(mx, sc[i * WPAD + j]);
        float s = 0.f;
        for (int j = 0; j < Ww; ++j) s += __expf(sc[i * WPAD + j] - mx);
        rmax[i] = mx; rsum[i] = s;
    }
    for (int j = threadIdx.x; j < Ww; j += 256) {
        float mx = -3.4e38f;
        for (int i = 0; i < Ww; ++i) mx = fmaxf(mx, sc[i * WPAD + j]);
        float s = 0.f;
        for (int i = 0; i < Ww; ++i) s += __expf(sc[i * WPAD + j] - mx);
        cmax[j] = mx; csum[j] = s;
    }
    __syncthreads();

    // ---- validity-mask sums: V_l2r[i]=sum_j M_l2r[j][i], V_r2l[j]=sum_i M_r2l[i][j]
    for (int i = threadIdx.x; i < Ww; i += 256) {
        float s = 0.f;
        for (int j = 0; j < Ww; ++j)
            s += __expf(sc[i * WPAD + j] - cmax[j]) / csum[j];
        mvalL[rowbase + i] = s;
        float s2 = 0.f;
        for (int k = 0; k < Ww; ++k)
            s2 += __expf(sc[k * WPAD + i] - rmax[k]) / rsum[k];
        mvalR[rowbase + i] = s2;
    }

    // ---- buf_l = M_r2l x xr,  buf_r = M_l2r x xl ----
    const _Float16* xrb = xrh + rowbase * Cch;
    const _Float16* xlb = xlh + rowbase * Cch;

    for (int t = wv; t < 160; t += 8) {          // 10 row-tiles x 16 col-tiles
        int mt = t / 16, nt = t % 16;
        int i0 = mt * 16, c0 = nt * 16;
        int irow = i0 + ln;
        float rm = rmax[irow], ri = 1.f / rsum[irow];
        v8f acc = {0.f,0.f,0.f,0.f,0.f,0.f,0.f,0.f};
#pragma unroll
        for (int kk = 0; kk < 5; ++kk) {         // K = 160 = 5*32
            int k0 = kk * 32;
            v16h a, bf;
            for (int e = 0; e < 16; ++e) {
                int j = k0 + (e & 7) + ((e >= 8) ? 16 : 0) + hl * 8;
                a[e]  = (_Float16)(__expf(sc[irow * WPAD + j] - rm) * ri);
                bf[e] = xrb[(long)j * Cch + c0 + ln];
            }
            acc = wmma16(a, bf, acc);
        }
        for (int r = 0; r < 8; ++r) {
            int m = r + hl * 8;
            bufL[(rowbase + i0 + m) * Cch + c0 + ln] = (_Float16)acc[r];
        }
    }

    for (int t = wv; t < 160; t += 8) {
        int mt = t / 16, nt = t % 16;
        int j0 = mt * 16, c0 = nt * 16;
        int jrow = j0 + ln;
        float cm = cmax[jrow], ci = 1.f / csum[jrow];
        v8f acc = {0.f,0.f,0.f,0.f,0.f,0.f,0.f,0.f};
#pragma unroll
        for (int kk = 0; kk < 5; ++kk) {
            int k0 = kk * 32;
            v16h a, bf;
            for (int e = 0; e < 16; ++e) {
                int i = k0 + (e & 7) + ((e >= 8) ? 16 : 0) + hl * 8;
                a[e]  = (_Float16)(__expf(sc[i * WPAD + jrow] - cm) * ci);
                bf[e] = xlb[(long)i * Cch + c0 + ln];
            }
            acc = wmma16(a, bf, acc);
        }
        for (int r = 0; r < 8; ++r) {
            int m = r + hl * 8;
            bufR[(rowbase + j0 + m) * Cch + c0 + ln] = (_Float16)acc[r];
        }
    }
}

// --------------------------- morphology, per image -------------------------

__global__ __launch_bounds__(256) void k_morph(const float* __restrict__ mval,
                                               float* __restrict__ Vout) {
    extern __shared__ char smem[];
    const int NP = Hh * Ww;
    int* lab = (int*)smem;                       // 25600 ints
    int* cnt = lab + NP;                         // 25600 ints
    unsigned char* mb  = (unsigned char*)(cnt + NP);   // 25600 B
    unsigned char* ext = mb + NP;                      // 166*166 B
    __shared__ int flag;

    int b = blockIdx.x;
    const float* mv = mval + (long)b * NP;

    for (int p = threadIdx.x; p < NP; p += 256)
        mb[p] = (mv[p] <= 0.1f) ? 1 : 0;
    __syncthreads();

    for (int pass = 0; pass < 2; ++pass) {
        for (int p = threadIdx.x; p < NP; p += 256) {
            bool on = pass == 0 ? (mb[p] != 0) : (mb[p] == 0);
            lab[p] = on ? p : -1;
        }
        __syncthreads();
        for (int it = 0; it < 4096; ++it) {      // monotone in-place propagation
            if (threadIdx.x == 0) flag = 0;
            __syncthreads();
            int changed = 0;
            for (int p = threadIdx.x; p < NP; p += 256) {
                if (lab[p] < 0) continue;
                int y = p / Ww, x = p % Ww;
                int mx = lab[p];
                for (int dy = -1; dy <= 1; ++dy)
                    for (int dx = -1; dx <= 1; ++dx) {
                        int yy = y + dy, xx = x + dx;
                        if ((unsigned)yy < (unsigned)Hh &&
                            (unsigned)xx < (unsigned)Ww) {
                            int v = lab[yy * Ww + xx];
                            if (v > mx) mx = v;
                        }
                    }
                if (mx != lab[p]) { lab[p] = mx; changed = 1; }
            }
            if (changed) flag = 1;
            __syncthreads();
            if (flag == 0) break;
            __syncthreads();
        }
        for (int p = threadIdx.x; p < NP; p += 256) cnt[p] = 0;
        __syncthreads();
        for (int p = threadIdx.x; p < NP; p += 256)
            if (lab[p] >= 0) atomicAdd(&cnt[lab[p]], 1);
        __syncthreads();
        for (int p = threadIdx.x; p < NP; p += 256) {
            if (pass == 0) { if (mb[p] && cnt[lab[p]] < 20)  mb[p] = 0; }
            else           { if (!mb[p] && cnt[lab[p]] <= 10) mb[p] = 1; }
        }
        __syncthreads();
    }

    const int E = Hh + 6;                        // closing on extended domain
    for (int q = threadIdx.x; q < E * E; q += 256) {
        int y = q / E - 3, x = q % E - 3;
        int v = 0;
        for (int dy = -3; dy <= 3 && !v; ++dy)
            for (int dx = -3; dx <= 3; ++dx) {
                if (dy * dy + dx * dx > 9) continue;
                int yy = y + dy, xx = x + dx;
                if ((unsigned)yy < (unsigned)Hh &&
                    (unsigned)xx < (unsigned)Ww && mb[yy * Ww + xx]) { v = 1; break; }
            }
        ext[q] = (unsigned char)v;
    }
    __syncthreads();
    float* Vo = Vout + (long)b * NP;
    for (int p = threadIdx.x; p < NP; p += 256) {
        int y = p / Ww, x = p % Ww;
        int all = 1;
        for (int dy = -3; dy <= 3 && all; ++dy)
            for (int dx = -3; dx <= 3; ++dx) {
                if (dy * dy + dx * dx > 9) continue;
                if (!ext[(y + 3 + dy) * E + (x + 3 + dx)]) { all = 0; break; }
            }
        Vo[p] = all ? 0.f : 1.f;                 // V = ~closed(m)
    }
}

// ------------- final 1x1 conv over gathered [buf | x | V] (K=544) ----------

__global__ __launch_bounds__(256) void k_gemm_out(const _Float16* __restrict__ bufX,
                                                  const _Float16* __restrict__ xh,
                                                  const float* __restrict__ Vm,
                                                  const _Float16* __restrict__ bnT,
                                                  const float* __restrict__ bias,
                                                  float* __restrict__ out,
                                                  int ocoff) {
    int lane = threadIdx.x & 31, wv = threadIdx.x >> 5;
    int tile = blockIdx.x * 8 + wv;
    int mt = tile >> 4, nt = tile & 15;
    long pbase = (long)mt * 16;
    int  ocbase = nt * 16;
    int  hl = lane >> 4, ln = lane & 15;

    long prow = pbase + ln;
    const _Float16* wrow = bnT + (long)(ocbase + ln) * KBN;
    v8f acc = {0.f,0.f,0.f,0.f,0.f,0.f,0.f,0.f};

#pragma unroll
    for (int kk = 0; kk < KBN / 32; ++kk) {
        int k0 = kk * 32;
        v16h a, bf;
        for (int half = 0; half < 2; ++half) {
            int ks = k0 + half * 16 + hl * 8;
            if (ks < 256) {
                v8h v = *(const v8h*)(bufX + prow * Cch + ks);
                for (int i = 0; i < 8; ++i) a[half * 8 + i] = v[i];
            } else if (ks < 512) {
                v8h v = *(const v8h*)(xh + prow * Cch + (ks - 256));
                for (int i = 0; i < 8; ++i) a[half * 8 + i] = v[i];
            } else {
                for (int i = 0; i < 8; ++i) {
                    int k = ks + i;
                    a[half * 8 + i] = (k == 512) ? (_Float16)Vm[prow]
                                                 : (_Float16)0.f;
                }
            }
            v8h bv = *(const v8h*)(wrow + ks);
            for (int i = 0; i < 8; ++i) bf[half * 8 + i] = bv[i];
        }
        acc = wmma16(a, bf, acc);
    }

    int oc = ocbase + ln;
    float bb = bias[oc];
    for (int r = 0; r < 8; ++r) {
        long p = pbase + r + hl * 8;
        int x = (int)(p % Ww);
        int y = (int)((p / Ww) % Hh);
        int bi = (int)(p / ((long)Ww * Hh));
        out[(((long)bi * 512 + ocoff + oc) * Hh + y) * Ww + x] = acc[r] + bb;
    }
}

// ------------------------------- launcher ----------------------------------

extern "C" void kernel_launch(void* const* d_in, const int* in_sizes, int n_in,
                              void* d_out, int out_size, void* d_ws, size_t ws_size,
                              hipStream_t stream) {
    const float* x_left  = (const float*)d_in[0];
    const float* x_right = (const float*)d_in[1];
    const float* rb_w1   = (const float*)d_in[2];
    const float* rb_w2   = (const float*)d_in[3];
    const float* b1_w    = (const float*)d_in[4];
    const float* b1_b    = (const float*)d_in[5];
    const float* b2_w    = (const float*)d_in[6];
    const float* b2_b    = (const float*)d_in[7];
    const float* bn_w    = (const float*)d_in[8];
    const float* bn_b    = (const float*)d_in[9];

    char* ws = (char*)d_ws;
    const size_t SZA = (size_t)NPX * Cch * sizeof(_Float16);           // 100 MB
    const size_t SZP = (size_t)Bsz * HP * WP * Cch * sizeof(_Float16); // 102.5 MB

    _Float16* xhL   = (_Float16*)(ws + 0 * SZA);
    _Float16* xhR   = (_Float16*)(ws + 1 * SZA);
    _Float16* blh   = (_Float16*)(ws + 2 * SZA);    // bl; later Sh
    _Float16* brh   = (_Float16*)(ws + 3 * SZA);    // br; later bufL
    _Float16* bufR  = (_Float16*)(ws + 4 * SZA);
    _Float16* xpadL = (_Float16*)(ws + 5 * SZA);
    _Float16* xpadR = (_Float16*)(ws + 5 * SZA + SZP);
    _Float16* y1pad = (_Float16*)(ws + 5 * SZA + 2 * SZP);  // conv temp; later Qh
    _Float16* Qh = y1pad;
    _Float16* Sh = blh;
    _Float16* bufL = brh;

    char* sm = ws + 5 * SZA + 3 * SZP;
    _Float16* w1T = (_Float16*)sm; sm += (size_t)K3 * Cch * 2;
    _Float16* w2T = (_Float16*)sm; sm += (size_t)K3 * Cch * 2;
    _Float16* b1T = (_Float16*)sm; sm += (size_t)Cch * Cch * 2;
    _Float16* b2T = (_Float16*)sm; sm += (size_t)Cch * Cch * 2;
    _Float16* bnT = (_Float16*)sm; sm += (size_t)Cch * KBN * 2;
    float* mvalL = (float*)sm; sm += (size_t)NPX * 4;
    float* mvalR = (float*)sm; sm += (size_t)NPX * 4;
    float* VL    = (float*)sm; sm += (size_t)NPX * 4;
    float* VR    = (float*)sm; sm += (size_t)NPX * 4;

    const dim3 blk(256);
    const int GEMM_BLOCKS = NPX / 8;             // per-wave 16x16 tile kernels
    const int CONV_BLOCKS = NPX / 32;            // 32px x 256oc blocks

    (void)hipFuncSetAttribute((const void*)k_attn,
        hipFuncAttributeMaxDynamicSharedMemorySize, 160 * WPAD * 4 + 4 * 160 * 4);
    (void)hipFuncSetAttribute((const void*)k_morph,
        hipFuncAttributeMaxDynamicSharedMemorySize, 258048);

    // zero halos (capture-safe async memsets), then pack activations + weights
    (void)hipMemsetAsync(xpadL, 0, SZP, stream);
    (void)hipMemsetAsync(xpadR, 0, SZP, stream);
    (void)hipMemsetAsync(y1pad, 0, SZP, stream);
    k_pack_x <<<(NPX * Cch) / 256, blk, 0, stream>>>(x_left,  xhL, xpadL);
    k_pack_x <<<(NPX * Cch) / 256, blk, 0, stream>>>(x_right, xhR, xpadR);
    k_pack_w3<<<(K3 * Cch) / 256,  blk, 0, stream>>>(rb_w1, w1T);
    k_pack_w3<<<(K3 * Cch) / 256,  blk, 0, stream>>>(rb_w2, w2T);
    k_pack_w1<<<(Cch * Cch) / 256, blk, 0, stream>>>(b1_w, b1T);
    k_pack_w1<<<(Cch * Cch) / 256, blk, 0, stream>>>(b2_w, b2T);
    k_pack_bn<<<(Cch * KBN) / 256, blk, 0, stream>>>(bn_w, bnT);

    // residual blocks: conv3-SiLU (padded out) then conv3 + skip (unpadded out)
    k_conv3<<<CONV_BLOCKS, blk, 0, stream>>>(xpadL, w1T, nullptr, y1pad, 0, 1);
    k_conv3<<<CONV_BLOCKS, blk, 0, stream>>>(y1pad, w2T, xhL,     blh,   1, 0);
    k_conv3<<<CONV_BLOCKS, blk, 0, stream>>>(xpadR, w1T, nullptr, y1pad, 0, 1);
    k_conv3<<<CONV_BLOCKS, blk, 0, stream>>>(y1pad, w2T, xhR,     brh,   1, 0);

    // Q = bl x b1_w^T (into dead y1pad), S = br x b2_w^T (into dead blh)
    k_gemm256<<<GEMM_BLOCKS, blk, 0, stream>>>(blh, b1T, b1_b, Qh);
    k_gemm256<<<GEMM_BLOCKS, blk, 0, stream>>>(brh, b2T, b2_b, Sh);

    // fused score / softmax / mask sums / attention-apply, one WG per (b,h)
    size_t attn_smem = (size_t)(160 * WPAD + 4 * 160) * sizeof(float);
    k_attn<<<Bsz * Hh, blk, attn_smem, stream>>>(Qh, Sh, xhL, xhR,
                                                 bufL, bufR, mvalL, mvalR);

    size_t morph_smem = 258048;
    k_morph<<<Bsz, blk, morph_smem, stream>>>(mvalL, VL);
    k_morph<<<Bsz, blk, morph_smem, stream>>>(mvalR, VR);

    float* outp = (float*)d_out;
    k_gemm_out<<<GEMM_BLOCKS, blk, 0, stream>>>(bufL, xhL, VL, bnT, bn_b, outp, 0);
    k_gemm_out<<<GEMM_BLOCKS, blk, 0, stream>>>(bufR, xhR, VR, bnT, bn_b, outp, 256);

    (void)in_sizes; (void)n_in; (void)out_size; (void)ws_size;
}
